// DINActivationUnit_32779190403191
// MI455X (gfx1250) — compile-verified, matrix operand order checked by reading, not run on hardware
//
#include <hip/hip_runtime.h>
#include <hip/hip_bf16.h>

typedef __attribute__((ext_vector_type(16))) __bf16 v16bf;
typedef __attribute__((ext_vector_type(8)))  float  v8f;

#define DD 128
#define TT 200
#define TP 208      // 13 tiles * 16
#define NTILE 13
#define TROW 136    // padded LDS row stride (bf16 elems), 272B = 16B/8B-aligned

// ---------------------------------------------------------------------------
// Prep: fold w1 blocks (h = (Wa+Wc)q + (Wb-Wc)k + Wd(q*k)).
//   WqT[f][n] = w1[n][f] + w1[n][256+f]                (f32, ws[0 .. 64KB))
//   Wk[n][f]  = w1[n][128+f] - w1[n][256+f]            (bf16 B-fragments)
//   Wp[n][f]  = w1[n][384+f]                           (bf16 B-fragments)
// Fragment layout: [m(2)][nblk(8)][kc(4)][lane(32)][j(16)] contiguous bf16,
// matching the 16-bit 16x32 operand register layout (lane<16 holds K {0..7,
// 16..23} of kc*32, lane>=16 holds K {8..15, 24..31}).
// ---------------------------------------------------------------------------
__global__ __launch_bounds__(256) void din_prep(const float* __restrict__ w1,
                                                float* __restrict__ wqT,
                                                __bf16* __restrict__ wfrag) {
  int idx = blockIdx.x * 256 + threadIdx.x;
  if (idx < DD * DD) {
    int n = idx & 127, f = idx >> 7;
    wqT[f * DD + n] = w1[n * 512 + f] + w1[n * 512 + 256 + f];
  } else {
    int e = idx - DD * DD;            // 0 .. 32767
    int j    = e & 15;
    int lane = (e >> 4) & 31;
    int kc   = (e >> 9) & 3;
    int nblk = (e >> 11) & 7;
    int m    = (e >> 14) & 1;
    int n  = nblk * 16 + (lane & 15);
    int hi = lane >> 4;
    int K  = kc * 32 + ((j < 8) ? (hi * 8 + j) : (16 + hi * 8 + (j - 8)));
    float v = (m == 0) ? (w1[n * 512 + 128 + K] - w1[n * 512 + 256 + K])
                       : w1[n * 512 + 384 + K];
    wfrag[e] = (__bf16)v;
  }
}

// ---------------------------------------------------------------------------
// Main: one block (8 waves) per batch row.
// ---------------------------------------------------------------------------
__global__ __launch_bounds__(256) void din_main(
    const float* __restrict__ query, const float* __restrict__ keys,
    const unsigned char* __restrict__ mask, const float* __restrict__ b1,
    const float* __restrict__ prelu_a, const float* __restrict__ w2,
    const float* __restrict__ b2, const float* __restrict__ wqT,
    const __bf16* __restrict__ wfrag, float* __restrict__ out) {
  __shared__ float  sQ[DD];
  __shared__ float  sW2[DD];
  __shared__ float  sU[DD];
  __shared__ float  sUp[2][DD];
  __shared__ float  sScore[TP];
  __shared__ float  sRed[DD];
  __shared__ float  sR8[8];
  __shared__ __bf16 kTile[16 * TROW];
  __shared__ __bf16 qkTile[16 * TROW];

  const int tid  = threadIdx.x;
  const int lane = tid & 31;
  const int wave = tid >> 5;          // wave owns N-columns [wave*16, wave*16+16)
  const int b    = blockIdx.x;
  const float* kb = keys + (size_t)b * TT * DD;
  const float pa  = prelu_a[0];
  const float b2v = b2[0];

  if (tid < DD) { sQ[tid] = query[b * DD + tid]; sW2[tid] = w2[tid]; }
  if (tid < TP) sScore[tid] = 0.f;
  __syncthreads();

  // u[n] = sum_f WqT[f][n] * q[f] + b1[n]   (coalesced over n)
  {
    int n = tid & 127, h = tid >> 7;
    float acc = 0.f;
    for (int f = h * 64; f < h * 64 + 64; ++f) acc += wqT[f * DD + n] * sQ[f];
    sUp[h][n] = acc;
  }
  __syncthreads();
  if (tid < DD) sU[tid] = sUp[0][tid] + sUp[1][tid] + b1[tid];

  // Persistent bf16 weight B-fragments for this wave's N-slice.
  v16bf wkf[4], wpf[4];
#pragma unroll
  for (int kc = 0; kc < 4; ++kc) {
    const uint4* g0 = (const uint4*)(wfrag + (size_t)((((0 * 8 + wave) * 4 + kc) * 32 + lane) * 16));
    const uint4* g1 = (const uint4*)(wfrag + (size_t)((((1 * 8 + wave) * 4 + kc) * 32 + lane) * 16));
    union { uint4 u[2]; v16bf v; } uk, up;
    uk.u[0] = g0[0]; uk.u[1] = g0[1];
    up.u[0] = g1[0]; up.u[1] = g1[1];
    wkf[kc] = uk.v;  wpf[kc] = up.v;
  }
  __syncthreads();   // sU ready

  const int   nn  = lane & 15;
  const int   hi  = lane >> 4;
  const float un  = sU[wave * 16 + nn];
  const float w2n = sW2[wave * 16 + nn];

  for (int tile = 0; tile < NTILE; ++tile) {
    const int t0 = tile * 16;
    // stage k and q*k tiles (bf16), float4-vectorized: 512 chunks of 4 elems
    for (int idx = tid; idx < 16 * 32; idx += 256) {
      int r = idx >> 5, c4 = (idx & 31) << 2;
      int gt = t0 + r;
      float4 kv = (gt < TT) ? *(const float4*)(kb + gt * DD + c4)
                            : make_float4(0.f, 0.f, 0.f, 0.f);
      float4 qv = *(const float4*)(sQ + c4);
      union { __bf16 h[4]; unsigned long long u; } pk, pq;
      pk.h[0] = (__bf16)kv.x;         pk.h[1] = (__bf16)kv.y;
      pk.h[2] = (__bf16)kv.z;         pk.h[3] = (__bf16)kv.w;
      pq.h[0] = (__bf16)(kv.x * qv.x); pq.h[1] = (__bf16)(kv.y * qv.y);
      pq.h[2] = (__bf16)(kv.z * qv.z); pq.h[3] = (__bf16)(kv.w * qv.w);
      *(unsigned long long*)(kTile  + r * TROW + c4) = pk.u;
      *(unsigned long long*)(qkTile + r * TROW + c4) = pq.u;
    }
    __syncthreads();

    // prefetch all 8 A-fragments, then run two independent WMMA chains
    v16bf a1f[4], a2f[4];
#pragma unroll
    for (int kc = 0; kc < 4; ++kc) {
      union { uint4 u[2]; v16bf v; } a1, a2;
      const uint4* p1 = (const uint4*)(kTile  + nn * TROW + kc * 32 + hi * 8);
      const uint4* p2 = (const uint4*)(qkTile + nn * TROW + kc * 32 + hi * 8);
      a1.u[0] = p1[0]; a1.u[1] = p1[2];   // K groups +0..7 and +16..23 (hi -> +8)
      a2.u[0] = p2[0]; a2.u[1] = p2[2];
      a1f[kc] = a1.v;  a2f[kc] = a2.v;
    }
    v8f acc1 = {}, acc2 = {};
#pragma unroll
    for (int kc = 0; kc < 4; ++kc) {
      acc1 = __builtin_amdgcn_wmma_f32_16x16x32_bf16(false, a1f[kc], false, wkf[kc],
                                                     (short)0, acc1, false, false);
      acc2 = __builtin_amdgcn_wmma_f32_16x16x32_bf16(false, a2f[kc], false, wpf[kc],
                                                     (short)0, acc2, false, false);
    }
    v8f acc = acc1 + acc2;

    // epilogue: h = acc + u ; PReLU ; dot with w2 over this wave's 16 columns
#pragma unroll
    for (int r = 0; r < 8; ++r) {
      float v = acc[r] + un;
      v = (v >= 0.f) ? v : pa * v;
      float s = v * w2n;
      s += __shfl_xor(s, 1, 32);
      s += __shfl_xor(s, 2, 32);
      s += __shfl_xor(s, 4, 32);
      s += __shfl_xor(s, 8, 32);          // sum within each 16-lane half
      if (nn == 0) atomicAdd(&sScore[t0 + r + hi * 8], s);
    }
    __syncthreads();
  }

  // ---- masked softmax + renormalization ----
  float mflag = 0.f;
  float sval  = -3.0e38f;
  if (tid < TT) {
    unsigned char m = mask[(size_t)b * TT + tid];
    mflag = m ? 1.f : 0.f;
    sval  = m ? (sScore[tid] + b2v) : -1e9f;
  }
  __syncthreads();

  float rmax = sval;
  rmax = fmaxf(rmax, __shfl_xor(rmax, 1, 32));
  rmax = fmaxf(rmax, __shfl_xor(rmax, 2, 32));
  rmax = fmaxf(rmax, __shfl_xor(rmax, 4, 32));
  rmax = fmaxf(rmax, __shfl_xor(rmax, 8, 32));
  rmax = fmaxf(rmax, __shfl_xor(rmax, 16, 32));
  if (lane == 0) sR8[wave] = rmax;
  __syncthreads();
  if (tid == 0) {
    float v = sR8[0];
#pragma unroll
    for (int i = 1; i < 8; ++i) v = fmaxf(v, sR8[i]);
    sRed[0] = v;
  }
  __syncthreads();
  const float mx = sRed[0];

  float p  = (tid < TT) ? expf(sval - mx) : 0.f;
  float sp = p;
  sp += __shfl_xor(sp, 1, 32);
  sp += __shfl_xor(sp, 2, 32);
  sp += __shfl_xor(sp, 4, 32);
  sp += __shfl_xor(sp, 8, 32);
  sp += __shfl_xor(sp, 16, 32);
  if (lane == 0) sR8[wave] = sp;
  __syncthreads();
  if (tid == 0) {
    float v = 0.f;
#pragma unroll
    for (int i = 0; i < 8; ++i) v += sR8[i];
    sRed[0] = v;
  }
  __syncthreads();
  const float denom = sRed[0];

  float wgt = (tid < TT) ? (p / denom) * mflag : 0.f;
  float sw  = wgt;
  sw += __shfl_xor(sw, 1, 32);
  sw += __shfl_xor(sw, 2, 32);
  sw += __shfl_xor(sw, 4, 32);
  sw += __shfl_xor(sw, 8, 32);
  sw += __shfl_xor(sw, 16, 32);
  if (lane == 0) sR8[wave] = sw;
  __syncthreads();
  if (tid == 0) {
    float v = 0.f;
#pragma unroll
    for (int i = 0; i < 8; ++i) v += sR8[i];
    sRed[0] = v;
  }
  __syncthreads();
  const float inv = 1.f / fmaxf(sRed[0], 1e-6f);
  if (tid < TT) sScore[tid] = wgt * inv;
  __syncthreads();

  // ---- out[b,:] = sum_t w[t] * keys[b,t,:]  (f32 keys re-read for accuracy)
  {
    int d = tid & 127, part = tid >> 7;
    float acc = 0.f;
    for (int t = part * 100; t < part * 100 + 100; ++t)
      acc += sScore[t] * kb[t * DD + d];
    if (part == 1) sRed[d] = acc;
    __syncthreads();
    if (part == 0) out[(size_t)b * DD + d] = acc + sRed[d];
  }
}

// ---------------------------------------------------------------------------
extern "C" void kernel_launch(void* const* d_in, const int* in_sizes, int n_in,
                              void* d_out, int out_size, void* d_ws, size_t ws_size,
                              hipStream_t stream) {
  const float*         query   = (const float*)d_in[0];
  const float*         keys    = (const float*)d_in[1];
  const unsigned char* mask    = (const unsigned char*)d_in[2];
  const float*         w1      = (const float*)d_in[3];
  const float*         b1      = (const float*)d_in[4];
  const float*         prelu_a = (const float*)d_in[5];
  const float*         w2      = (const float*)d_in[6];
  const float*         b2      = (const float*)d_in[7];
  float*               out     = (float*)d_out;

  float*  wqT   = (float*)d_ws;                                     // 64 KB
  __bf16* wfrag = (__bf16*)((char*)d_ws + DD * DD * sizeof(float)); // 64 KB

  const int B = in_sizes[0] / DD;

  din_prep<<<(DD * DD + 2 * DD * DD) / 256, 256, 0, stream>>>(w1, wqT, wfrag);
  din_main<<<B, 256, 0, stream>>>(query, keys, mask, b1, prelu_a, w2, b2,
                                  wqT, wfrag, out);
}
